// PWCStereo_20280835572149
// MI455X (gfx1250) — matrix operand order repaired and checked by reading, not compiled, not run on hardware
//
#include <hip/hip_runtime.h>

typedef __bf16 bf16;
typedef __attribute__((ext_vector_type(16))) __bf16 v16bf;
typedef __attribute__((ext_vector_type(8)))  __bf16 v8bf;
typedef __attribute__((ext_vector_type(8)))  float  v8f;

#define LRELU_SLOPE 0.1f

static inline int ceil_div_h(int a, int b) { return (a + b - 1) / b; }
static inline int pad32_h(int c) { return (c + 31) & ~31; }

// ---------------- layout / conversion kernels ----------------

// OIHW f32 weights -> [CoutPad][9][CinPad] bf16, zero-padded
__global__ void k_repack_w(const float* __restrict__ w, bf16* __restrict__ d,
                           int Cin, int Cout, int Cpad, int CoutPad) {
    int i = blockIdx.x * blockDim.x + threadIdx.x;
    int total = CoutPad * 9 * Cpad;
    if (i >= total) return;
    int n   = i / (9 * Cpad);
    int rem = i - n * 9 * Cpad;
    int tap = rem / Cpad;
    int c   = rem - tap * Cpad;
    bf16 v = (bf16)0.0f;
    if (n < Cout && c < Cin) v = (bf16)w[((size_t)n * Cin + c) * 9 + tap];
    d[i] = v;
}

// planar C x HW f32 -> NHWC bf16 with channel stride Cpad (zero pad)
__global__ void k_nchw_f32_to_nhwc_bf16(const float* __restrict__ src, bf16* __restrict__ dst,
                                        int C, int Cpad, int HW) {
    int i = blockIdx.x * blockDim.x + threadIdx.x;
    if (i >= HW * Cpad) return;
    int p = i / Cpad, c = i - p * Cpad;
    dst[i] = (c < C) ? (bf16)src[(size_t)c * HW + p] : (bf16)0.0f;
}

// zero channels [from, to) of an NHWC buffer
__global__ void k_zero_ch(bf16* __restrict__ dst, int dS, int from, int to, int HW) {
    int n = (to - from);
    int i = blockIdx.x * blockDim.x + threadIdx.x;
    if (i >= n * HW) return;
    int p = i / n, c = i - p * n;
    dst[(size_t)p * dS + from + c] = (bf16)0.0f;
}

// strided NHWC channel-block copy (for concatenation)
__global__ void k_copy_ch(bf16* __restrict__ dst, int dS, int dOff,
                          const bf16* __restrict__ src, int sS, int sOff,
                          int C, int HW) {
    int i = blockIdx.x * blockDim.x + threadIdx.x;
    if (i >= C * HW) return;
    int p = i / C, c = i - p * C;
    dst[(size_t)p * dS + dOff + c] = src[(size_t)p * sS + sOff + c];
}

// planar 1-channel f32 -> one NHWC channel slot
__global__ void k_f32c_to_bf16_ch(const float* __restrict__ s, bf16* __restrict__ d,
                                  int dS, int off, int HW) {
    int i = blockIdx.x * blockDim.x + threadIdx.x;
    if (i < HW) d[(size_t)i * dS + off] = (bf16)s[i];
}

// ---------------- flow arithmetic kernels (planar f32) ----------------

__global__ void k_add_inplace(float* __restrict__ d, const float* __restrict__ s, int n) {
    int i = blockIdx.x * blockDim.x + threadIdx.x;
    if (i < n) d[i] += s[i];
}

__global__ void k_clamp_flow(float* __restrict__ d, int n, int neg) {
    int i = blockIdx.x * blockDim.x + threadIdx.x;
    if (i < n) {
        float v = d[i];
        d[i] = neg ? fminf(v, 0.f) : fmaxf(v, 0.f);
    }
}

__global__ void k_upsample1(const float* __restrict__ in, float* __restrict__ out,
                            int H, int W, int f, float scale, float addc) {
    int Ho = H * f, Wo = W * f;
    int i = blockIdx.x * blockDim.x + threadIdx.x;
    if (i >= Ho * Wo) return;
    int oy = i / Wo, ox = i % Wo;
    float sy = (Ho > 1) ? oy * ((float)(H - 1) / (float)(Ho - 1)) : 0.f;
    float sx = (Wo > 1) ? ox * ((float)(W - 1) / (float)(Wo - 1)) : 0.f;
    int y0 = (int)sy, x0 = (int)sx;
    int y1 = y0 + 1 < H ? y0 + 1 : H - 1;
    int x1 = x0 + 1 < W ? x0 + 1 : W - 1;
    float wy = sy - (float)y0, wx = sx - (float)x0;
    float v = (in[y0 * W + x0] * (1.f - wx) + in[y0 * W + x1] * wx) * (1.f - wy)
            + (in[y1 * W + x0] * (1.f - wx) + in[y1 * W + x1] * wx) * wy;
    out[i] = v * scale + addc;
}

// ---------------- NHWC warp / correlation ----------------

// horizontal-only flow warp; x/out NHWC with channel stride Cs
__global__ void k_warp_x(const bf16* __restrict__ x, const float* __restrict__ flow,
                         bf16* __restrict__ out, int C, int Cs, int H, int W) {
    int p = blockIdx.x * blockDim.x + threadIdx.x;
    if (p >= H * W) return;
    int y = p / W, xx = p % W;
    float px = (float)xx + flow[p];
    float fx = floorf(px);
    int x0 = (int)fx;
    float wx = px - fx;
    int x1 = x0 + 1;
    bool v0 = (x0 >= 0) && (x0 <= W - 1);
    bool v1 = (x1 >= 0) && (x1 <= W - 1);
    int xc0 = x0 < 0 ? 0 : (x0 > W - 1 ? W - 1 : x0);
    int xc1 = x1 < 0 ? 0 : (x1 > W - 1 ? W - 1 : x1);
    const bf16* r0 = x + ((size_t)y * W + xc0) * Cs;
    const bf16* r1 = x + ((size_t)y * W + xc1) * Cs;
    bf16* o = out + (size_t)p * Cs;
    for (int c = 0; c < C; ++c) {
        float g0 = v0 ? (float)r0[c] : 0.f;
        float g1 = v1 ? (float)r1[c] : 0.f;
        o[c] = (bf16)(g0 * (1.f - wx) + g1 * wx);
    }
}

// 81-tap cost volume, NHWC in (stride Cs), bf16 NHWC out (stride dS, channels 0..80)
__global__ void k_corr(const bf16* __restrict__ a, const bf16* __restrict__ b,
                       bf16* __restrict__ dst, int dS,
                       int Cs, int C, int H, int W, int act) {
    const int md = 4, D = 9;
    int p = blockIdx.x * blockDim.x + threadIdx.x;
    if (p >= H * W) return;
    int y = p / W, x = p % W;
    float inv = 1.f / (float)C;
    const bf16* ap = a + (size_t)p * Cs;
    bf16* o = dst + (size_t)p * dS;
    for (int dy = 0; dy < D; ++dy) {
        int yy = y + dy - md;
        for (int dx = 0; dx < D; ++dx) {
            int xx = x + dx - md;
            float s = 0.f;
            if ((unsigned)yy < (unsigned)H && (unsigned)xx < (unsigned)W) {
                const bf16* bp = b + ((size_t)yy * W + xx) * Cs;
                for (int c = 0; c < C; ++c)
                    s += (float)ap[c] * (float)bp[c];
            }
            s *= inv;
            if (act) s = s > 0.f ? s : LRELU_SLOPE * s;
            o[dy * D + dx] = (bf16)s;
        }
    }
}

// ---------------- conv3x3 implicit GEMM, bf16 WMMA, NHWC ----------------
// K ordered (tap, channel), channels padded to mult-of-32: every chunk full.
// A fragment = 2x global_load_b128 (lane-private, straight from global).
// B tile: DOUBLE-BUFFERED LDS stage via global_load_async_to_lds_b128; the
// copy for chunk t+1 overlaps the ds_load+WMMA of chunk t; one barrier/chunk.
// B fragment = 2x ds_load_b128. Tile: 64 spatial x 32 outch, 8 wave32.

#define CONV_MT 64
#define CONV_NT 32

__global__ __launch_bounds__(256)
void k_conv3x3_wmma(const bf16* __restrict__ in, const bf16* __restrict__ wgt,
                    const float* __restrict__ bias,
                    float* __restrict__ outF, bf16* __restrict__ outB,
                    int Cpad, int H, int W, int Cout, int CoutPad, int Ho, int Wo,
                    int stride, int dil, int act) {
    __shared__ __align__(32) bf16 sB[2][2][32][16];   // [buf][nt][frag lane][elem], 4 KB

    const int tid  = threadIdx.x;
    const int lane = tid & 31;
    const int wv   = tid >> 5;
    const int wm   = wv & 3;
    const int wn   = wv >> 2;
    const int m0   = blockIdx.x * CONV_MT;
    const int n0   = blockIdx.y * CONV_NT;
    const int HoWo = Ho * Wo;
    const int pad  = dil;

    // fixed per-lane output pixel
    const int mrow = m0 + wm * 16 + (lane & 15);
    const bool mvalid = (mrow < HoWo);
    const int oy = mvalid ? (mrow / Wo) : 0;
    const int ox = mvalid ? (mrow - oy * Wo) : 0;
    const int by = oy * stride - pad;
    const int bx = ox * stride - pad;

    // B staging: threads 0..127 each copy 16 contiguous bytes per chunk
    const int st_nt   = (tid >> 6) & 1;
    const int st_half = (tid >> 5) & 1;
    const int st_fl   = tid & 31;
    const int st_n    = n0 + st_nt * 16 + (st_fl & 15);
    const int st_koff = ((st_fl < 16) ? 0 : 8) + st_half * 16;
    const unsigned st_lds0 = (unsigned)(size_t)&sB[0][st_nt][st_fl][st_half * 8];
    const unsigned st_lds1 = (unsigned)(size_t)&sB[1][st_nt][st_fl][st_half * 8];
    const bf16* wrow = wgt + (size_t)st_n * 9 * Cpad;

    const int koff = (lane < 16) ? 0 : 8;   // A fragment channel sub-offset

    v8f acc = {0.f, 0.f, 0.f, 0.f, 0.f, 0.f, 0.f, 0.f};

    // staging cursor (uniform across block)
    int s_tap = 0, s_c0 = 0, s_buf = 0;
    auto stageNext = [&]() {
        if (s_tap < 9) {
            if (tid < 128) {
                const bf16* src = wrow + (size_t)s_tap * Cpad + s_c0 + st_koff;
                unsigned dst = s_buf ? st_lds1 : st_lds0;
                asm volatile("global_load_async_to_lds_b128 %0, %1, off"
                             :: "v"(dst), "v"(src) : "memory");
            }
            s_buf ^= 1;
            s_c0 += 32;
            if (s_c0 >= Cpad) { s_c0 = 0; ++s_tap; }
        }
    };

    stageNext();          // prologue: chunk 0 -> buffer 0
    int rbuf = 0;

    for (int tap = 0; tap < 9; ++tap) {
        const int ky = tap / 3, kx = tap - ky * 3;
        const int iy = by + ky * dil;
        const int ix = bx + kx * dil;
        const bool valid = mvalid && (unsigned)iy < (unsigned)H && (unsigned)ix < (unsigned)W;
        const size_t q = valid ? ((size_t)iy * W + ix) : 0;
        const bf16* arow = in + q * Cpad;
        const bf16* wtap = wrow + (size_t)tap * Cpad;

        for (int c0 = 0; c0 < Cpad; c0 += 32) {
            if (tid == 0)
                __builtin_prefetch(wtap + c0 + Cpad, 0, 1);   // next tap, same chunk

            // A fragment: two contiguous 8-channel runs (2x global_load_b128)
            v8bf lo = {}, hi = {};
            if (valid) {
                const bf16* ab = arow + c0 + koff;
                lo = *(const v8bf*)ab;
                hi = *(const v8bf*)(ab + 16);
            }
            v16bf a = __builtin_shufflevector(lo, hi,
                0, 1, 2, 3, 4, 5, 6, 7, 8, 9, 10, 11, 12, 13, 14, 15);

            // current chunk's stage (issued last iteration) must have landed
            asm volatile("s_wait_asynccnt 0x0" ::: "memory");
            __syncthreads();

            // overlap: kick off the copy for the NEXT chunk into the other buffer
            stageNext();

            v16bf b = *(const v16bf*)(&sB[rbuf][wn][lane][0]);   // 2x ds_load_b128
            rbuf ^= 1;
            acc = __builtin_amdgcn_wmma_f32_16x16x32_bf16(
                false, a, false, b, (short)0, acc, false, false);
        }
    }

    // epilogue: bias + optional leaky-relu; zero-fill padded out channels
    int n = n0 + wn * 16 + (lane & 15);
    #pragma unroll
    for (int r = 0; r < 8; ++r) {
        int m = m0 + wm * 16 + r + ((lane >> 4) << 3);
        if (m < HoWo) {
            float v = 0.f;
            if (n < Cout) {
                v = acc[r] + bias[n];
                if (act) v = v > 0.f ? v : LRELU_SLOPE * v;
            }
            if (outB) outB[(size_t)m * CoutPad + n] = (bf16)v;
            if (outF && n < Cout) outF[(size_t)n * HoWo + m] = v;
        }
    }
}

// ---------------- host orchestration ----------------

extern "C" void kernel_launch(void* const* d_in, const int* in_sizes, int n_in,
                              void* d_out, int out_size, void* d_ws, size_t ws_size,
                              hipStream_t stream) {
    (void)in_sizes; (void)n_in; (void)out_size; (void)ws_size;

    char* cur = (char*)d_ws;
    auto alloc = [&](size_t bytes) -> void* {
        void* r = (void*)cur;
        cur += (bytes + 255) & ~(size_t)255;
        return r;
    };

    struct Layer { const float* w; const float* b; bf16* wb; int ci, cip, co, cop; };
    int ii = 1;
    auto takeLayer = [&](int ci, int co) -> Layer {
        Layer L;
        L.w = (const float*)d_in[ii++];
        L.b = (const float*)d_in[ii++];
        L.ci = ci; L.cip = pad32_h(ci); L.co = co; L.cop = pad32_h(co);
        int n = L.cop * 9 * L.cip;
        L.wb = (bf16*)alloc((size_t)n * sizeof(bf16));
        k_repack_w<<<ceil_div_h(n, 256), 256, 0, stream>>>(L.w, L.wb, ci, co, L.cip, L.cop);
        return L;
    };

    static const int LV[7] = {3, 16, 32, 64, 96, 128, 192};
    Layer pyr[6][2];
    for (int l = 0; l < 6; ++l) {
        pyr[l][0] = takeLayer(LV[l], LV[l + 1]);
        pyr[l][1] = takeLayer(LV[l + 1], LV[l + 1]);
    }
    static const int REV[5] = {192, 128, 96, 64, 32};
    static const int EO[5]  = {128, 128, 96, 64, 32};
    Layer est[5][6];
    for (int i = 0; i < 5; ++i) {
        int c = 81 + (i > 0 ? REV[i] + 1 : 0);
        for (int j = 0; j < 5; ++j) { est[i][j] = takeLayer(c, EO[j]); c = EO[j]; }
        est[i][5] = takeLayer(32, 1);
    }
    static const int CO[7] = {128, 128, 128, 96, 64, 32, 1};
    static const int CD[7] = {1, 2, 4, 8, 16, 1, 1};
    Layer ctxL[7];
    { int c = 33; for (int j = 0; j < 7; ++j) { ctxL[j] = takeLayer(c, CO[j]); c = CO[j]; } }

    auto conv = [&](const bf16* in, const Layer& L, float* outF, bf16* outB,
                    int H, int W, int stride, int dil, int act) {
        int Ho = (H - 1) / stride + 1, Wo = (W - 1) / stride + 1;
        dim3 g(ceil_div_h(Ho * Wo, CONV_MT), L.cop / CONV_NT);
        k_conv3x3_wmma<<<g, 256, 0, stream>>>(in, L.wb, L.b, outF, outB,
                                              L.cip, H, W, L.co, L.cop, Ho, Wo,
                                              stride, dil, act);
    };

    // ---- input images -> NHWC bf16 (Cpad = 32) ----
    const float* xin = (const float*)d_in[0];
    const int IH = 256, IW = 832, IHW = IH * IW;
    bf16* im1 = (bf16*)alloc((size_t)32 * IHW * sizeof(bf16));
    bf16* im2 = (bf16*)alloc((size_t)32 * IHW * sizeof(bf16));
    k_nchw_f32_to_nhwc_bf16<<<ceil_div_h(32 * IHW, 256), 256, 0, stream>>>(xin, im1, 3, 32, IHW);
    k_nchw_f32_to_nhwc_bf16<<<ceil_div_h(32 * IHW, 256), 256, 0, stream>>>(
        xin + (size_t)3 * IHW, im2, 3, 32, IHW);

    // ---- feature pyramids (NHWC, stride pad32(ch)) ----
    static const int PH[6] = {128, 64, 32, 16, 8, 4};
    static const int PWd[6] = {416, 208, 104, 52, 26, 13};
    bf16* f1[6]; bf16* f2[6];
    for (int l = 0; l < 6; ++l) {
        size_t sz = (size_t)pad32_h(LV[l + 1]) * PH[l] * PWd[l] * sizeof(bf16);
        f1[l] = (bf16*)alloc(sz);
        f2[l] = (bf16*)alloc(sz);
    }
    bf16* ptmp = (bf16*)alloc((size_t)32 * 128 * 416 * sizeof(bf16));

    auto pyramid = [&](const bf16* im, bf16* const* F) {
        const bf16* c_in = im;
        int H = IH, Wd = IW;
        for (int l = 0; l < 6; ++l) {
            conv(c_in, pyr[l][0], nullptr, ptmp, H, Wd, 2, 1, 1);
            conv(ptmp, pyr[l][1], nullptr, F[l], PH[l], PWd[l], 1, 1, 1);
            c_in = F[l]; H = PH[l]; Wd = PWd[l];
        }
    };
    pyramid(im1, f1);
    pyramid(im2, f2);

    // ---- PWC working buffers (sized for largest level i=4: 64x208) ----
    const int MAXHW = 64 * 208;
    bf16*  catB   = (bf16*)alloc((size_t)224 * MAXHW * sizeof(bf16));
    bf16*  warpB  = (bf16*)alloc((size_t)128 * MAXHW * sizeof(bf16));
    bf16*  tA     = (bf16*)alloc((size_t)128 * MAXHW * sizeof(bf16));
    bf16*  tB     = (bf16*)alloc((size_t)128 * MAXHW * sizeof(bf16));
    bf16*  ctxCat = (bf16*)alloc((size_t)64 * MAXHW * sizeof(bf16));
    float* flow   = (float*)alloc((size_t)MAXHW * sizeof(float));
    float* upf    = (float*)alloc((size_t)MAXHW * sizeof(float));
    float* ctxF   = (float*)alloc((size_t)MAXHW * sizeof(float));

    static const size_t OUT_OFF[5] = {0, 425984, 532480, 559104, 565760};

    auto pwc = [&](bf16* const* X1, bf16* const* X2, int neg) {
        static const int LH[5] = {4, 8, 16, 32, 64};
        static const int LW[5] = {13, 26, 52, 104, 208};
        static const int LC[5] = {192, 128, 96, 64, 32};
        for (int i = 0; i < 5; ++i) {
            int h = LH[i], w = LW[i], hw = h * w, ch = LC[i];
            int xS = pad32_h(ch);             // feature channel stride
            int catS = est[i][0].cip;         // concat buffer channel stride
            const bf16* x1 = X1[5 - i];
            const bf16* x2 = X2[5 - i];
            if (i == 0) {
                k_corr<<<ceil_div_h(hw, 256), 256, 0, stream>>>(
                    x1, x2, catB, catS, xS, ch, h, w, 0);
                k_zero_ch<<<ceil_div_h((catS - 81) * hw, 256), 256, 0, stream>>>(
                    catB, catS, 81, catS, hw);
            } else {
                k_warp_x<<<ceil_div_h(hw, 256), 256, 0, stream>>>(x2, upf, warpB, ch, xS, h, w);
                k_corr<<<ceil_div_h(hw, 256), 256, 0, stream>>>(
                    x1, warpB, catB, catS, xS, ch, h, w, 1);
                k_copy_ch<<<ceil_div_h(ch * hw, 256), 256, 0, stream>>>(
                    catB, catS, 81, x1, xS, 0, ch, hw);
                k_f32c_to_bf16_ch<<<ceil_div_h(hw, 256), 256, 0, stream>>>(
                    upf, catB, catS, 81 + ch, hw);
                int cin = 81 + ch + 1;
                if (catS > cin)
                    k_zero_ch<<<ceil_div_h((catS - cin) * hw, 256), 256, 0, stream>>>(
                        catB, catS, cin, catS, hw);
            }
            // flow estimator
            const bf16* c_in = catB;
            bf16* ob = tA;
            for (int j = 0; j < 5; ++j) {
                conv(c_in, est[i][j], nullptr, ob, h, w, 1, 1, 1);
                c_in = ob;
                ob = (ob == tA) ? tB : tA;
            }
            const bf16* feat = c_in;   // 32ch NHWC (stride 32)
            conv(feat, est[i][5], flow, nullptr, h, w, 1, 1, 0);
            if (i > 0) k_add_inplace<<<ceil_div_h(hw, 256), 256, 0, stream>>>(flow, upf, hw);
            k_clamp_flow<<<ceil_div_h(hw, 256), 256, 0, stream>>>(flow, hw, neg);
            if (i == 4) {
                // context net on concat([flow, feat]) -> 33ch, stride 64
                k_f32c_to_bf16_ch<<<ceil_div_h(hw, 256), 256, 0, stream>>>(
                    flow, ctxCat, 64, 0, hw);
                k_copy_ch<<<ceil_div_h(32 * hw, 256), 256, 0, stream>>>(
                    ctxCat, 64, 1, feat, 32, 0, 32, hw);
                k_zero_ch<<<ceil_div_h(31 * hw, 256), 256, 0, stream>>>(
                    ctxCat, 64, 33, 64, hw);
                const bf16* ci2 = ctxCat;
                bf16* ob2 = tA;
                for (int j = 0; j < 6; ++j) {
                    conv(ci2, ctxL[j], nullptr, ob2, h, w, 1, CD[j], 1);
                    ci2 = ob2;
                    ob2 = (ob2 == tA) ? tB : tA;
                }
                conv(ci2, ctxL[6], ctxF, nullptr, h, w, 1, 1, 0);
                k_add_inplace<<<ceil_div_h(hw, 256), 256, 0, stream>>>(flow, ctxF, hw);
                k_clamp_flow<<<ceil_div_h(hw, 256), 256, 0, stream>>>(flow, hw, neg);
            }
            // final output: upsample(flow*4, 4) (+1e-6), rl direction negated
            int j = 4 - i;
            float* dst = (float*)d_out + OUT_OFF[j] + (size_t)neg * 16 * hw;
            k_upsample1<<<ceil_div_h(16 * hw, 256), 256, 0, stream>>>(
                flow, dst, h, w, 4, neg ? -4.f : 4.f, 1e-6f);
            if (i < 4)
                k_upsample1<<<ceil_div_h(4 * hw, 256), 256, 0, stream>>>(
                    flow, upf, h, w, 2, 1.f, 0.f);
        }
    };

    pwc(f1, f2, 0);  // left->right
    pwc(f2, f1, 1);  // right->left (neg)
}
